// Attention_32916629357041
// MI455X (gfx1250) — matrix operand order repaired and checked by reading, not compiled, not run on hardware
//
#include <hip/hip_runtime.h>

typedef __bf16 bf16;
typedef __attribute__((ext_vector_type(16))) __bf16 v16bf;
typedef __attribute__((ext_vector_type(8)))  __bf16 v8bf;
typedef __attribute__((ext_vector_type(8)))  float  v8f;

#define NWIN   1024
#define NTOK   64
#define DIM    512
#define HEADS  8
#define DH     64
#define INNER  512
#define ATT_SCALE 0.125f

#define ROWS 128  // rows (tokens) per block = 2 windows
#define XP 520    // LDS pitch (halves) for 512-wide bf16 rows (16B-aligned rows, conflict-free)
#define QP 72     // LDS pitch (halves) for 64-wide bf16 rows
#define SP 68     // LDS pitch (floats) for 64-wide f32 rows

// A-frag for v_wmma_f32_16x16x32_bf16: lane L holds A[m0+(L&15)][kb..kb+7] and [kb+16..kb+23],
// kb = kc + 8*(L>>4). Source is row-major [rows][pitch] bf16.
__device__ __forceinline__ v16bf load_frag_a(const bf16* base, int pitch, int m0, int kc, int lane) {
  const bf16* p = base + (m0 + (lane & 15)) * pitch + kc + 8 * (lane >> 4);
  union { v16bf v; v8bf h[2]; } u;
  u.h[0] = *(const v8bf*)(p);
  u.h[1] = *(const v8bf*)(p + 16);
  return u.v;
}
// B-frag (B = W^T, W row-major [n][K]): lane L holds W[n0+(L&15)][kb..kb+7] and [kb+16..kb+23], kb = kc + 8*(L>>4).
__device__ __forceinline__ v16bf load_frag_b(const bf16* base, int stride, int n0, int kc, int lane) {
  const bf16* p = base + (n0 + (lane & 15)) * stride + kc + 8 * (lane >> 4);
  union { v16bf v; v8bf h[2]; } u;
  u.h[0] = *(const v8bf*)(p);
  u.h[1] = *(const v8bf*)(p + 16);
  return u.v;
}
#define WMMA_BF16(a, b, c) \
  __builtin_amdgcn_wmma_f32_16x16x32_bf16(false, (a), false, (b), (short)0, (c), false, false)

// ---------------- Kernel 0: weight bf16 conversion + relative-position bias table ----------------
__global__ void prep_kernel(const float* __restrict__ wqkv, const float* __restrict__ wout,
                            const float* __restrict__ Mat,
                            bf16* __restrict__ wq_bf, bf16* __restrict__ wo_bf,
                            float* __restrict__ bias_tab) {
  int i = blockIdx.x * blockDim.x + threadIdx.x;
  const int NWQ = 3 * INNER * DIM;   // 786432
  const int NWO = DIM * INNER;       // 262144
  if (i < NWQ) { wq_bf[i] = (bf16)wqkv[i]; return; }
  i -= NWQ;
  if (i < NWO) { wo_bf[i] = (bf16)wout[i]; return; }
  i -= NWO;
  if (i < NTOK * NTOK) {
    int qi = i >> 6, kj = i & 63;
    int ri = qi >> 3, ci = qi & 7, rj = kj >> 3, cj = kj & 7;
    bias_tab[i] = Mat[(ri - rj + 7) * 15 + (ci - cj + 7)];
  }
}

// ---------------- Kernel 1: fused qkv projection + attention (2 windows per block) ----------------
// 8 waves; wave w: n-tile = 16*(w&3); 4 m-tiles = 64*(w>>2) + {0,16,32,48} (one window per wave).
// 240 KB LDS => 1 workgroup/WGP => 2 waves/SIMD => allow up to 512 VGPRs/wave (no spills).
__global__ void __launch_bounds__(256)
__attribute__((amdgpu_waves_per_eu(2)))
attn_kernel(const float* __restrict__ x,
            const bf16* __restrict__ wq_bf,
            const float* __restrict__ bias_tab,
            bf16* __restrict__ aout) {
  extern __shared__ char smem[];
  bf16*  xs    = (bf16*)smem;                      // [128][XP] x rows (2 windows), bf16
  bf16*  qs    = xs + ROWS * XP;                   // [128][QP] q (pre-scaled); reused as attn
  bf16*  ksm   = qs + ROWS * QP;                   // [128][QP] k
  bf16*  vT    = ksm + ROWS * QP;                  // [2][64][QP] v transposed: [win][dim][token]
  float* simb  = (float*)(vT + ROWS * QP);         // [128][SP] logits f32
  float* biasS = (float*)(simb + ROWS * SP);       // [64][64]  bias tile

  const int tid  = threadIdx.x;
  const int lane = tid & 31;
  const int w    = tid >> 5;
  const int wb   = blockIdx.x;                     // window pair

  // Stage 128 token rows (f32 -> bf16) and bias tile
  const float* xw = x + (size_t)wb * ROWS * DIM;
  for (int i = tid; i < ROWS * DIM; i += 256)
    xs[(i >> 9) * XP + (i & 511)] = (bf16)xw[i];
  for (int i = tid; i < NTOK * NTOK; i += 256)
    biasS[i] = bias_tab[i];
  __syncthreads();

  const int n0    = 16 * (w & 3);
  const int g     = w >> 2;           // which window this wave's m-tiles live in
  const int mbase = 64 * g;
  const int nn    = n0 + (lane & 15);
  const int half  = lane >> 4;

  #pragma unroll 1
  for (int h = 0; h < HEADS; ++h) {
    // --- q, k, v projections: [128x64] = xs[128x512] @ W_head[64x512]^T ---
    #pragma unroll 1
    for (int t = 0; t < 3; ++t) {
      const bf16* Wrow = wq_bf + (size_t)(t * INNER + h * DH) * DIM;
      v8f acc[4] = {};
      #pragma unroll
      for (int kc = 0; kc < DIM; kc += 32) {
        v16bf bfrag = load_frag_b(Wrow, DIM, n0, kc, lane);
        v16bf af[4];
        #pragma unroll
        for (int i = 0; i < 4; ++i)
          af[i] = load_frag_a(xs, XP, mbase + 16 * i, kc, lane);
        #pragma unroll
        for (int i = 0; i < 4; ++i)
          acc[i] = WMMA_BF16(af[i], bfrag, acc[i]);
      }
      if (t == 0) {
        #pragma unroll
        for (int i = 0; i < 4; ++i) {
          bf16* p = qs + (mbase + 16 * i + 8 * half) * QP + nn;
          #pragma unroll
          for (int r = 0; r < 8; ++r) p[r * QP] = (bf16)(acc[i][r] * ATT_SCALE);
        }
      } else if (t == 1) {
        #pragma unroll
        for (int i = 0; i < 4; ++i) {
          bf16* p = ksm + (mbase + 16 * i + 8 * half) * QP + nn;
          #pragma unroll
          for (int r = 0; r < 8; ++r) p[r * QP] = (bf16)acc[i][r];
        }
      } else {  // store v transposed per window: vT[g][dim][token]
        #pragma unroll
        for (int i = 0; i < 4; ++i) {
          bf16* p = vT + ((size_t)g * 64 + nn) * QP + (16 * i + 8 * half);
          #pragma unroll
          for (int r = 0; r < 8; ++r) p[r] = (bf16)acc[i][r];
        }
      }
    }
    __syncthreads();

    // --- sim = q @ k^T within each window ---
    {
      v8f acc[4] = {};
      #pragma unroll
      for (int kc = 0; kc < DH; kc += 32) {
        v16bf bfrag = load_frag_b(ksm + (size_t)mbase * QP, QP, n0, kc, lane);
        v16bf af[4];
        #pragma unroll
        for (int i = 0; i < 4; ++i)
          af[i] = load_frag_a(qs, QP, mbase + 16 * i, kc, lane);
        #pragma unroll
        for (int i = 0; i < 4; ++i)
          acc[i] = WMMA_BF16(af[i], bfrag, acc[i]);
      }
      #pragma unroll
      for (int i = 0; i < 4; ++i) {
        float* p = simb + (mbase + 16 * i + 8 * half) * SP + nn;
        #pragma unroll
        for (int r = 0; r < 8; ++r) p[r * SP] = acc[i][r];
      }
    }
    __syncthreads();

    // --- softmax over 64 keys (bias added here); 2 lanes per row, shfl_xor(1) reduce ---
    // attn is written into qs (q is dead after sim).
    {
      const int row = tid >> 1;            // 0..127 (global row)
      const int c0  = (tid & 1) * 32;
      float vv[32];
      float mx = -1e30f;
      #pragma unroll
      for (int j = 0; j < 32; ++j) {
        vv[j] = simb[row * SP + c0 + j] + biasS[(row & 63) * 64 + c0 + j];
        mx = fmaxf(mx, vv[j]);
      }
      mx = fmaxf(mx, __shfl_xor(mx, 1));
      float s = 0.f;
      #pragma unroll
      for (int j = 0; j < 32; ++j) { vv[j] = __expf(vv[j] - mx); s += vv[j]; }
      s += __shfl_xor(s, 1);
      const float inv = 1.0f / s;
      #pragma unroll
      for (int j = 0; j < 32; ++j)
        qs[row * QP + c0 + j] = (bf16)(vv[j] * inv);
    }
    __syncthreads();

    // --- out_h = attn @ v (B-frags contiguous from vT[g]) ---
    {
      v8f acc[4] = {};
      #pragma unroll
      for (int kc = 0; kc < NTOK; kc += 32) {
        v16bf bfrag = load_frag_b(vT + (size_t)g * 64 * QP, QP, n0, kc, lane);
        v16bf af[4];
        #pragma unroll
        for (int i = 0; i < 4; ++i)
          af[i] = load_frag_a(qs, QP, mbase + 16 * i, kc, lane);
        #pragma unroll
        for (int i = 0; i < 4; ++i)
          acc[i] = WMMA_BF16(af[i], bfrag, acc[i]);
      }
      bf16* dst = aout + (size_t)wb * ROWS * INNER + h * DH;
      #pragma unroll
      for (int i = 0; i < 4; ++i) {
        bf16* p = dst + (size_t)(mbase + 16 * i + 8 * half) * INNER + nn;
        #pragma unroll
        for (int r = 0; r < 8; ++r) p[(size_t)r * INNER] = (bf16)acc[i][r];
      }
    }
    __syncthreads();
  }
}

// ---------------- Kernel 2: output projection  out = aout @ w_out^T + b_out ----------------
// 128-row blocks, wave = 4 m-tiles x 1 n-tile; 133 KB LDS => 2 blocks/WGP => 4 waves/SIMD.
__global__ void __launch_bounds__(256)
__attribute__((amdgpu_waves_per_eu(4)))
proj_kernel(const bf16* __restrict__ aout,
            const bf16* __restrict__ wo_bf,
            const float* __restrict__ b_out,
            float* __restrict__ out) {
  extern __shared__ char smem[];
  bf16* as = (bf16*)smem;  // [128][XP]
  const int tid  = threadIdx.x;
  const int lane = tid & 31;
  const int w    = tid >> 5;
  const size_t mblk = (size_t)blockIdx.y * ROWS;

  // Stage 128x512 bf16 A-strip (copy as dwords)
  const unsigned int* s32 = (const unsigned int*)(aout + mblk * INNER);
  unsigned int* d32 = (unsigned int*)as;
  for (int i = tid; i < ROWS * INNER / 2; i += 256)
    d32[(i >> 8) * (XP / 2) + (i & 255)] = s32[i];
  __syncthreads();

  const int n0    = 16 * (w & 3);
  const int mbase = 64 * (w >> 2);
  const int half  = lane >> 4;
  const int col   = blockIdx.x * 64 + n0 + (lane & 15);
  const bf16* Wrow = wo_bf + (size_t)(blockIdx.x * 64) * INNER;

  v8f acc[4] = {};
  #pragma unroll
  for (int kc = 0; kc < INNER; kc += 32) {
    v16bf bfrag = load_frag_b(Wrow, INNER, n0, kc, lane);
    v16bf af[4];
    #pragma unroll
    for (int i = 0; i < 4; ++i)
      af[i] = load_frag_a(as, XP, mbase + 16 * i, kc, lane);
    #pragma unroll
    for (int i = 0; i < 4; ++i)
      acc[i] = WMMA_BF16(af[i], bfrag, acc[i]);
  }
  const float bias = b_out[col];
  #pragma unroll
  for (int i = 0; i < 4; ++i) {
    float* p = out + (mblk + mbase + 16 * i + 8 * half) * DIM + col;
    #pragma unroll
    for (int r = 0; r < 8; ++r) p[(size_t)r * DIM] = acc[i][r] + bias;
  }
}

extern "C" void kernel_launch(void* const* d_in, const int* in_sizes, int n_in,
                              void* d_out, int out_size, void* d_ws, size_t ws_size,
                              hipStream_t stream) {
  const float* x     = (const float*)d_in[0];
  const float* w_qkv = (const float*)d_in[1];
  const float* w_out = (const float*)d_in[2];
  const float* b_out = (const float*)d_in[3];
  const float* Mat   = (const float*)d_in[4];
  float* out = (float*)d_out;

  char* ws = (char*)d_ws;
  bf16*  wq_bf    = (bf16*)(ws + 0);              // 1,572,864 B
  bf16*  wo_bf    = (bf16*)(ws + 1572864);        //   524,288 B
  float* bias_tab = (float*)(ws + 2097152);       //    16,384 B
  bf16*  aout     = (bf16*)(ws + 2113536);        // 67,108,864 B (bf16 [1024*64, 512])

  {
    const int total = 3 * INNER * DIM + DIM * INNER + NTOK * NTOK;  // 1,052,672
    prep_kernel<<<(total + 255) / 256, 256, 0, stream>>>(w_qkv, w_out, Mat, wq_bf, wo_bf, bias_tab);
  }
  {
    const size_t lds = (size_t)ROWS * XP * 2      // xs        133,120
                     + 2 * (size_t)ROWS * QP * 2  // qs, ksm    36,864
                     + (size_t)ROWS * QP * 2      // vT         18,432
                     + (size_t)ROWS * SP * 4      // sim        34,816
                     + (size_t)NTOK * NTOK * 4;   // bias       16,384  => 239,616 B (< 320 KB/WGP)
    attn_kernel<<<NWIN / 2, 256, lds, stream>>>(x, wq_bf, bias_tab, aout);
  }
  {
    const size_t lds = (size_t)ROWS * XP * 2;     // 133,120 B (2 blocks/WGP)
    proj_kernel<<<dim3(DIM / 64, (NWIN * NTOK) / ROWS), 256, lds, stream>>>(aout, wo_bf, b_out, out);
  }
}
